// KA_GNN_two_60430189854855
// MI455X (gfx1250) — matrix operand order repaired and checked by reading, not compiled, not run on hardware
//
#include <hip/hip_runtime.h>
#include <hip/hip_bf16.h>
#include <hip/hip_fp16.h>

// ---------------------------------------------------------------------------
// KA-GNN (Fourier-KAN GNN) forward on gfx1250.
//
// Strategy: hoist the per-edge KAN transform to per-node (16x less compute),
// run all node-level KAN projections as f16 WMMA GEMMs with f32 accumulation,
// and do the edge aggregation as an L2-resident float atomic scatter.
// ---------------------------------------------------------------------------

typedef __attribute__((ext_vector_type(16))) _Float16 v16h;
typedef __attribute__((ext_vector_type(8)))  _Float16 v8h;
typedef __attribute__((ext_vector_type(8)))  float    v8f;

#define NN       50000      // nodes
#define NE       800000     // edges
#define INF      64         // input features
#define HID      32         // hidden
#define GRID_N   4
#define NGRAPH   128
#define NEG_SLOPE 0.01f

// feature index convention: f = i*(2*GRID) + s*GRID + g,  s=0 -> cos, s=1 -> sin
// so per input channel: [cos(1x)..cos(4x), sin(1x)..sin(4x)]  (8 features)

// ---------------------------------------------------------------------------
// Weight prep: f32 (2, out, in, grid) -> f16 transposed [K][32], K = in*8.
// ---------------------------------------------------------------------------
__global__ void prep_weights(const float* __restrict__ W_in,    // (2,32,64,4)
                             const float* __restrict__ W_conv,  // (2,2,32,32,4)
                             _Float16* __restrict__ Wt_in,      // [512][32]
                             _Float16* __restrict__ Wt_conv) {  // [2][256][32]
    int idx = blockIdx.x * blockDim.x + threadIdx.x;
    const int IN_ELE = 512 * 32;
    const int CV_ELE = 256 * 32;
    if (idx < IN_ELE) {
        int f = idx >> 5, o = idx & 31;
        int i = f >> 3, r = f & 7, s = r >> 2, g = r & 3;
        Wt_in[idx] = (_Float16)W_in[((s * 32 + o) * 64 + i) * 4 + g];
    } else if (idx < IN_ELE + 2 * CV_ELE) {
        int j  = idx - IN_ELE;
        int l  = j / CV_ELE;
        int jj = j % CV_ELE;
        int f = jj >> 5, o = jj & 31;
        int i = f >> 3, r = f & 7, s = r >> 2, g = r & 3;
        Wt_conv[j] = (_Float16)W_conv[(((l * 2 + s) * 32 + o) * 32 + i) * 4 + g];
    }
}

// ---------------------------------------------------------------------------
// Node-level Fourier-KAN projection:  out[N,32] = Fourier(X[N,IN_CH]) @ Wt
//   Kd = IN_CH*8.  Block = 128 threads (4 waves), 32 node rows per block.
//   Wave w -> output tile (rowTile = w>>1, colTile = w&1), 16x16 f32 tile,
//   accumulated over Kd with v_wmma_f32_16x16x32_f16.
// ---------------------------------------------------------------------------
template <int IN_CH>
__global__ void __launch_bounds__(128)
node_kan(const float* __restrict__ X,        // [nNodes, IN_CH]
         const _Float16* __restrict__ Wt,    // [Kd][32]
         float* __restrict__ out,            // [nNodes, 32]
         int nNodes) {
    constexpr int Kd = IN_CH * 8;
    constexpr int MB = 32;                   // node rows per block
    __shared__ _Float16 feat[MB * Kd];       // 32KB (IN=64) / 16KB (IN=32)

    const int nodeBase = blockIdx.x * MB;
    const int tid = threadIdx.x;

    // ---- stage Fourier features into LDS (uniform trip count: EXEC stays full)
    for (int p = tid; p < MB * IN_CH; p += 128) {
        int row = p / IN_CH, i = p % IN_CH;
        int node = nodeBase + row;
        if (node >= nNodes) node = nNodes - 1;   // clamp (padding rows harmless)
        float v = X[node * IN_CH + i];
        _Float16* fp = &feat[row * Kd + i * 8];
#pragma unroll
        for (int g = 0; g < GRID_N; ++g) {
            float s, c;
            __sincosf(v * (float)(g + 1), &s, &c);
            fp[g]          = (_Float16)c;
            fp[GRID_N + g] = (_Float16)s;
        }
    }
    __syncthreads();

    const int wave    = tid >> 5;
    const int lane    = tid & 31;
    const int rowTile = wave >> 1;           // 0..1
    const int colTile = wave & 1;            // 0..1
    const int m       = lane & 15;
    const int half    = lane >> 4;           // 0: lanes 0-15, 1: lanes 16-31
    const int abase   = half * 8;            // A-layout K sub-offset

    const _Float16* arow = &feat[(rowTile * 16 + m) * Kd];

    v8f acc = {};
#pragma unroll 4
    for (int kc = 0; kc < Kd; kc += 32) {
        // A fragment: 16-bit A 16x32 layout -> two contiguous 16B LDS loads
        v8h lo = *(const v8h*)(arow + kc + abase);
        v8h hi = *(const v8h*)(arow + kc + 16 + abase);
        v16h a = __builtin_shufflevector(lo, hi, 0, 1, 2, 3, 4, 5, 6, 7,
                                                 8, 9, 10, 11, 12, 13, 14, 15);
        // B fragment: lane holds N = lane&15, K in same order as A (abase+s, 16+abase+s)
        v16h b;
#pragma unroll
        for (int s = 0; s < 8; ++s) {
            b[s]     = Wt[(kc + abase + s) * 32 + colTile * 16 + m];
            b[8 + s] = Wt[(kc + 16 + abase + s) * 32 + colTile * 16 + m];
        }
        acc = __builtin_amdgcn_wmma_f32_16x16x32_f16(
                  false, a, false, b, (short)0, acc, false, false);
    }

    // ---- store C/D tile (VGPR r: M = r + 8*half, N = lane&15)
#pragma unroll
    for (int r = 0; r < 8; ++r) {
        int node = nodeBase + rowTile * 16 + r + half * 8;
        if (node < nNodes)
            out[node * 32 + colTile * 16 + (lane & 15)] = acc[r];
    }
}

// ---------------------------------------------------------------------------
// Edge aggregation: m[dst] += t[src]  (one thread per edge x 4-channel quad)
// ---------------------------------------------------------------------------
__global__ void edge_scatter(const float* __restrict__ t,
                             const int* __restrict__ src,
                             const int* __restrict__ dst,
                             float* __restrict__ msg, int nEdges) {
    int idx = blockIdx.x * blockDim.x + threadIdx.x;
    int e  = idx >> 3;
    int c4 = (idx & 7) * 4;
    if (e >= nEdges) return;
    int s = src[e];
    int d = dst[e];
    const float4 v = *(const float4*)(t + s * 32 + c4);
    float* p = msg + d * 32 + c4;
    atomicAdd(p + 0, v.x);
    atomicAdd(p + 1, v.y);
    atomicAdd(p + 2, v.z);
    atomicAdd(p + 3, v.w);
}

__global__ void fill_zero(float* __restrict__ p, int n) {
    int i = blockIdx.x * blockDim.x + threadIdx.x;
    if (i < n) p[i] = 0.0f;
}

__global__ void residual_lrelu(float* __restrict__ h,
                               const float* __restrict__ msg, int n) {
    int i = blockIdx.x * blockDim.x + threadIdx.x;
    if (i < n) {
        float v = msg[i] + h[i];
        h[i] = v > 0.0f ? v : NEG_SLOPE * v;
    }
}

// ---------------------------------------------------------------------------
// Mean pool per graph (batch is sorted, but atomics are cheap & simple here)
// ---------------------------------------------------------------------------
__global__ void pool_scatter(const float* __restrict__ h,
                             const int* __restrict__ batch,
                             float* __restrict__ sums,    // [NGRAPH*32]
                             float* __restrict__ counts,  // [NGRAPH]
                             int nNodes) {
    int idx = blockIdx.x * blockDim.x + threadIdx.x;
    if (idx >= nNodes * 32) return;
    int node = idx >> 5;
    int c    = idx & 31;
    int g    = batch[node];
    atomicAdd(&sums[g * 32 + c], h[idx]);
    if (c == 0) atomicAdd(&counts[g], 1.0f);
}

// readout: grid=1 KAN + bias + sigmoid
__global__ void readout(const float* __restrict__ sums,
                        const float* __restrict__ counts,
                        const float* __restrict__ W_out,   // (2,1,32,1) flat
                        const float* __restrict__ b_out,   // scalar
                        float* __restrict__ out) {
    int b = blockIdx.x * blockDim.x + threadIdx.x;
    if (b >= NGRAPH) return;
    float cnt = fmaxf(counts[b], 1.0f);
    float accum = 0.0f;
#pragma unroll
    for (int i = 0; i < HID; ++i) {
        float y = sums[b * 32 + i] / cnt;
        float s, c;
        __sincosf(y, &s, &c);
        accum += c * W_out[i] + s * W_out[32 + i];
    }
    float z = accum + b_out[0];
    out[b] = 1.0f / (1.0f + expf(-z));
}

// ---------------------------------------------------------------------------
// Launch
// ---------------------------------------------------------------------------
extern "C" void kernel_launch(void* const* d_in, const int* in_sizes, int n_in,
                              void* d_out, int out_size, void* d_ws, size_t ws_size,
                              hipStream_t stream) {
    const float* x        = (const float*)d_in[0];   // (50000, 64)
    const int*   eidx     = (const int*)  d_in[1];   // (2, 800000)
    const int*   batch    = (const int*)  d_in[2];   // (50000,)
    const float* W_in     = (const float*)d_in[3];   // (2, 32, 64, 4)
    const float* W_conv   = (const float*)d_in[4];   // (2, 2, 32, 32, 4)
    const float* W_out    = (const float*)d_in[5];   // (2, 1, 32, 1)
    const float* b_out    = (const float*)d_in[6];   // (1, 1)
    float*       out      = (float*)d_out;           // (128, 1)

    const int* src = eidx;
    const int* dst = eidx + NE;

    // workspace layout (bytes, all 256B-aligned by construction)
    char* ws = (char*)d_ws;
    float*    h       = (float*)   (ws + 0);                       // 6.4 MB
    float*    t       = (float*)   (ws + 6400000);                 // 6.4 MB
    float*    msg     = (float*)   (ws + 12800000);                // 6.4 MB
    _Float16* Wt_in   = (_Float16*)(ws + 19200000);                // 32 KB
    _Float16* Wt_conv = (_Float16*)(ws + 19232768);                // 32 KB
    float*    sums    = (float*)   (ws + 19265536);                // 16 KB
    float*    counts  = (float*)   (ws + 19281920);                // 512 B
    (void)in_sizes; (void)n_in; (void)out_size; (void)ws_size;

    // 1) weights -> f16, transposed [K][32]
    prep_weights<<<(32768 + 255) / 256, 256, 0, stream>>>(W_in, W_conv, Wt_in, Wt_conv);

    // 2) input KAN projection (K = 512) -> h
    node_kan<INF><<<(NN + 31) / 32, 128, 0, stream>>>(x, Wt_in, h, NN);

    const int nodeElems = NN * HID;           // 1.6M
    // 3) conv layers (message KAN hoisted to nodes: t = kan(h); m = scatter(t[src]))
    for (int l = 0; l < 2; ++l) {
        node_kan<HID><<<(NN + 31) / 32, 128, 0, stream>>>(h, Wt_conv + l * 256 * 32, t, NN);
        fill_zero<<<(nodeElems + 255) / 256, 256, 0, stream>>>(msg, nodeElems);
        edge_scatter<<<(NE * 8 + 255) / 256, 256, 0, stream>>>(t, src, dst, msg, NE);
        residual_lrelu<<<(nodeElems + 255) / 256, 256, 0, stream>>>(h, msg, nodeElems);
    }

    // 4) mean pool + readout
    fill_zero<<<(NGRAPH * 32 + NGRAPH + 255) / 256, 256, 0, stream>>>(sums, NGRAPH * 32 + NGRAPH);
    pool_scatter<<<(nodeElems + 255) / 256, 256, 0, stream>>>(h, batch, sums, counts, NN);
    readout<<<1, NGRAPH, 0, stream>>>(sums, counts, W_out, b_out, out);
}
